// RNNScratch_77300821393465
// MI455X (gfx1250) — matrix operand order repaired and checked by reading, not compiled, not run on hardware
//
#include <hip/hip_runtime.h>
#include <hip/hip_bf16.h>
#include <math.h>

// Problem dims (fixed by the reference)
#define T_STEPS 2048
#define BATCH   32
#define IN      512
#define HID     512
#define KTILES  16   // 512 / 32 (bf16 WMMA K)

typedef __attribute__((ext_vector_type(16))) __bf16 v16bf;
typedef __attribute__((ext_vector_type(8)))  __bf16 v8bf;
typedef __attribute__((ext_vector_type(8)))  float  v8f;
typedef int gcc_v4i __attribute__((vector_size(16)));

// ---------------------------------------------------------------- tanh ------
__device__ __forceinline__ float dev_tanh(float x) {
#if __has_builtin(__builtin_amdgcn_tanhf)
  return __builtin_amdgcn_tanhf(x);          // v_tanh_f32 (gfx1250 trans op)
#else
  return tanhf(x);
#endif
}

// ------------------------------------------- async global->LDS (ASYNCcnt) --
// Signature confirmed by round-1 diagnostic + round-2 clean compile:
//   (AS(1) int4*, AS(3) int4*, imm offset, cpol)
#if __has_builtin(__builtin_amdgcn_global_load_async_to_lds_b128)
#define HAVE_ASYNC_LDS 1
__device__ __forceinline__ void async_copy_b128(const float* g, float* l) {
  __builtin_amdgcn_global_load_async_to_lds_b128(
      (__attribute__((address_space(1))) gcc_v4i*)g,
      (__attribute__((address_space(3))) gcc_v4i*)l, 0, 0);
}
__device__ __forceinline__ void wait_async0() {
#if __has_builtin(__builtin_amdgcn_s_wait_asynccnt)
  __builtin_amdgcn_s_wait_asynccnt(0);
#else
  asm volatile("s_wait_asynccnt 0" ::: "memory");
#endif
}
#else
#define HAVE_ASYNC_LDS 0
__device__ __forceinline__ void async_copy_b128(const float* g, float* l) {
  *(float4*)l = *(const float4*)g;           // synchronous fallback
}
__device__ __forceinline__ void wait_async0() {}
#endif

// -------------------------------------------------------------------------
// Phase 0: pack W_xh and W_hh (fp32 row-major [K=512][N=512]) into bf16
// fragment-major layout: for n-tile j, k-block kb, lane L: 16 contiguous bf16
//   B[kb*32 + (L>>4)*16 + i][j*16 + (L&15)]   , i = 0..15
// => every lane's B fragment is one aligned 32-byte run (2x b128).
// -------------------------------------------------------------------------
__global__ void pack_weights(const float* __restrict__ Wxh,
                             const float* __restrict__ Whh,
                             __bf16* __restrict__ pXh,
                             __bf16* __restrict__ pHh) {
  int idx = blockIdx.x * blockDim.x + threadIdx.x;   // 0 .. 2*512*512-1
  const float* W = Wxh;
  __bf16* P = pXh;
  int e = idx;
  if (e >= HID * IN) { e -= HID * IN; W = Whh; P = pHh; }
  int i  = e & 15;
  int L  = (e >> 4) & 31;
  int kb = (e >> 9) & 15;
  int j  = e >> 13;
  int k  = kb * 32 + ((L >> 4) << 4) + i;
  int n  = j * 16 + (L & 15);
  P[e] = (__bf16)W[(size_t)k * HID + n];
}

__device__ __forceinline__ v16bf cvt_afrag(float4 u0, float4 u1,
                                           float4 u2, float4 u3) {
  v16bf a;
  a[0]=(__bf16)u0.x; a[1]=(__bf16)u0.y; a[2]=(__bf16)u0.z; a[3]=(__bf16)u0.w;
  a[4]=(__bf16)u1.x; a[5]=(__bf16)u1.y; a[6]=(__bf16)u1.z; a[7]=(__bf16)u1.w;
  a[8]=(__bf16)u2.x; a[9]=(__bf16)u2.y; a[10]=(__bf16)u2.z; a[11]=(__bf16)u2.w;
  a[12]=(__bf16)u3.x; a[13]=(__bf16)u3.y; a[14]=(__bf16)u3.z; a[15]=(__bf16)u3.w;
  return a;
}

// -------------------------------------------------------------------------
// Phase 1: xw = x @ W_xh + b_h  ->  out[0 .. T*B*H)
// One wave computes a 16x128 strip (8 n-tiles); fully unrolled K loop with
// one-iteration-ahead register prefetch of A (raw fp32) and all 8 B frags.
// -------------------------------------------------------------------------
__global__ __launch_bounds__(128) void input_proj(
    const float*  __restrict__ x,
    const __bf16* __restrict__ pXh,
    const float*  __restrict__ bh,
    float*        __restrict__ out) {
  const int lane   = threadIdx.x & 31;
  const int wave   = threadIdx.x >> 5;       // 0..3 -> n-group
  const int mtile  = blockIdx.x;             // 0..4095
  const int c      = (lane >> 4) << 4;       // A K-run base: 0 or 16
  const float* ap  = x + (size_t)(mtile * 16 + (lane & 15)) * IN;
  const int ntile0 = wave * 8;
  // fragment (ntile, kb) lives at pXh + ntile*8192 + kb*512 + lane*16
  const __bf16* wp = pXh + (size_t)lane * 16;

  v8f acc[8];
#pragma unroll
  for (int q = 0; q < 8; ++q) acc[q] = (v8f)0.0f;

  // prologue: kb = 0
  const float4* r1 = (const float4*)(ap + c);
  float4 u0 = r1[0], u1 = r1[1], u2 = r1[2], u3 = r1[3];
  v16bf bfr[8];
#pragma unroll
  for (int q = 0; q < 8; ++q)
    bfr[q] = *(const v16bf*)(wp + (size_t)(ntile0 + q) * 8192);

#pragma unroll
  for (int kb = 0; kb < KTILES; ++kb) {
    v16bf a = cvt_afrag(u0, u1, u2, u3);
    float4 n0{}, n1{}, n2{}, n3{};
    v16bf bn[8];
    if (kb + 1 < KTILES) {   // prefetch next iteration's operands
      const float4* p = (const float4*)(ap + (kb + 1) * 32 + c);
      n0 = p[0]; n1 = p[1]; n2 = p[2]; n3 = p[3];
#pragma unroll
      for (int q = 0; q < 8; ++q)
        bn[q] = *(const v16bf*)(wp + (size_t)(ntile0 + q) * 8192 +
                                (size_t)(kb + 1) * 512);
    }
#pragma unroll
    for (int q = 0; q < 8; ++q)
      acc[q] = __builtin_amdgcn_wmma_f32_16x16x32_bf16(
          false, a, false, bfr[q], (short)0, acc[q], false, false);
    u0 = n0; u1 = n1; u2 = n2; u3 = n3;
#pragma unroll
    for (int q = 0; q < 8; ++q) bfr[q] = bn[q];
  }

  const int nlow  = lane & 15;
  const int rbase = (lane >> 4) << 3;
#pragma unroll
  for (int q = 0; q < 8; ++q) {
    const int col = (ntile0 + q) * 16 + nlow;
    const float bias = bh[col];
#pragma unroll
    for (int r = 0; r < 8; ++r) {
      const int row = mtile * 16 + rbase + r;
      out[(size_t)row * HID + col] = acc[q][r] + bias;
    }
  }
}

// -------------------------------------------------------------------------
// Phase 2: persistent single-workgroup scan.
// 1024 threads = 32 waves; wave w owns n-tile w (16 cols) x both m-tiles.
// h kept as bf16 image in LDS (double buffered); W_hh fragments streamed
// from L2 with a depth-2 register pipeline; xw_t staged via async-to-LDS
// (ASYNCcnt), so it never perturbs the WMMA pipeline's LOADcnt waits.
// -------------------------------------------------------------------------
__global__ __launch_bounds__(1024) void rnn_scan(
    const __bf16* __restrict__ pHh,
    float* out) {                       // out[t] holds xw_t in, h_t out
  __shared__ __bf16 hbuf[2][BATCH * HID];   // 64 KB
  __shared__ float  xwbuf[BATCH * HID];     // 64 KB staging for xw_t

  const int tid   = threadIdx.x;
  const int lane  = tid & 31;
  const int ntile = tid >> 5;               // wave = n-tile (0..31)
  const int nlow  = lane & 15;
  const int hi    = lane >> 4;              // 0/1
  const int c     = hi << 4;                // A K-run base
  const int col   = ntile * 16 + nlow;      // D column for this lane
  // W_hh fragment (ntile, kb) at pHh + ntile*8192 + kb*512 + lane*16
  const __bf16* wp = pHh + (size_t)ntile * 8192 + (size_t)lane * 16;

  for (int t = 0; t < T_STEPS; ++t) {
    // kick off async stage of xw_t (this thread copies 64 contiguous bytes)
    {
      const float* g = out + (size_t)t * BATCH * HID + tid * 16;
      float* l = xwbuf + tid * 16;
#pragma unroll
      for (int q = 0; q < 4; ++q) async_copy_b128(g + q * 4, l + q * 4);
    }

    v8f acc0 = (v8f)0.0f, acc1 = (v8f)0.0f;
    if (t > 0) {
      const __bf16* hprev = hbuf[t & 1];
      v16bf b0 = *(const v16bf*)(wp);
      v16bf b1 = *(const v16bf*)(wp + 512);
#pragma unroll
      for (int kb = 0; kb < KTILES; ++kb) {
        v16bf bn;
        if (kb + 2 < KTILES)
          bn = *(const v16bf*)(wp + (size_t)(kb + 2) * 512);
        const __bf16* p0 = hprev + (size_t)nlow * HID + kb * 32 + c;
        v8bf a0lo = *(const v8bf*)(p0);
        v8bf a0hi = *(const v8bf*)(p0 + 8);
        v8bf a1lo = *(const v8bf*)(p0 + 16 * HID);
        v8bf a1hi = *(const v8bf*)(p0 + 16 * HID + 8);
        v16bf a0, a1;
#pragma unroll
        for (int i = 0; i < 8; ++i) {
          a0[i] = a0lo[i]; a0[8 + i] = a0hi[i];
          a1[i] = a1lo[i]; a1[8 + i] = a1hi[i];
        }
        acc0 = __builtin_amdgcn_wmma_f32_16x16x32_bf16(
            false, a0, false, b0, (short)0, acc0, false, false);
        acc1 = __builtin_amdgcn_wmma_f32_16x16x32_bf16(
            false, a1, false, b0, (short)0, acc1, false, false);
        b0 = b1; b1 = bn;
      }
    }

    // xw_t fully staged in LDS (own asyncs done + everyone arrived)
    wait_async0();
    __syncthreads();

    float h0[8], h1[8];
#pragma unroll
    for (int r = 0; r < 8; ++r) {
      float xw0 = xwbuf[(size_t)(hi * 8 + r) * HID + col];
      float xw1 = xwbuf[(size_t)(16 + hi * 8 + r) * HID + col];
      h0[r] = dev_tanh(acc0[r] + xw0);
      h1[r] = dev_tanh(acc1[r] + xw1);
    }

    // fp32 h_t -> out[t]
    float* po = out + (size_t)t * BATCH * HID;
#pragma unroll
    for (int r = 0; r < 8; ++r) {
      po[(size_t)(hi * 8 + r) * HID + col]      = h0[r];
      po[(size_t)(16 + hi * 8 + r) * HID + col] = h1[r];
    }
    if (t == T_STEPS - 1) {   // h_final output slot
      float* pf = out + (size_t)T_STEPS * BATCH * HID;
#pragma unroll
      for (int r = 0; r < 8; ++r) {
        pf[(size_t)(hi * 8 + r) * HID + col]      = h0[r];
        pf[(size_t)(16 + hi * 8 + r) * HID + col] = h1[r];
      }
    }

    // bf16 h_t -> LDS buffer for next step's A fragments
    __bf16* hb = hbuf[(t + 1) & 1];
#pragma unroll
    for (int r = 0; r < 8; ++r) {
      hb[(size_t)(hi * 8 + r) * HID + col]      = (__bf16)h0[r];
      hb[(size_t)(16 + hi * 8 + r) * HID + col] = (__bf16)h1[r];
    }

    __syncthreads();   // h_t visible; xwbuf free for next step's asyncs
  }
}

// -------------------------------------------------------------------------
extern "C" void kernel_launch(void* const* d_in, const int* in_sizes, int n_in,
                              void* d_out, int out_size, void* d_ws, size_t ws_size,
                              hipStream_t stream) {
  const float* x   = (const float*)d_in[0];   // [T,B,I]
  const float* Wxh = (const float*)d_in[1];   // [I,H]
  const float* Whh = (const float*)d_in[2];   // [H,H]
  const float* bh  = (const float*)d_in[3];   // [H]
  float* out = (float*)d_out;                 // [T,B,H] ++ [B,H]

  __bf16* pXh = (__bf16*)d_ws;                // 512 KB packed W_xh
  __bf16* pHh = pXh + (size_t)IN * HID;       // 512 KB packed W_hh

  // Phase 0: pack both weight matrices into bf16 fragment-major layout
  pack_weights<<<(2 * IN * HID) / 256, 256, 0, stream>>>(Wxh, Whh, pXh, pHh);

  // Phase 1: xw = x @ W_xh + b  -> out[0 .. T*B*H)
  input_proj<<<(T_STEPS * BATCH) / 16, 128, 0, stream>>>(x, pXh, bh, out);

  // Phase 2: persistent single-workgroup recurrent scan
  rnn_scan<<<1, 1024, 0, stream>>>(pHh, out);
}
